// ConbimambaBlock_91319594648177
// MI455X (gfx1250) — matrix operand order repaired and checked by reading, not compiled, not run on hardware
//
#include <hip/hip_runtime.h>
#include <hip/hip_bf16.h>
#include <cstdint>
#include <cstddef>

#define DIMD 512
#define DII  1024
#define NST  16
#define DTRR 32
#define KCONV 4
#define BB   8
#define LLEN 512
#define MTOK (BB*LLEN)   // 4096 tokens

typedef __bf16 bf16_t;
typedef __attribute__((ext_vector_type(16))) __bf16 bf16x16;
typedef __attribute__((ext_vector_type(8)))  float  f32x8;

union Frag {
    unsigned int u[8];
    bf16x16 v;
};

__device__ __forceinline__ float sigmoidf_(float x) { return 1.0f / (1.0f + __expf(-x)); }
__device__ __forceinline__ float siluf_(float x)    { return x * sigmoidf_(x); }
__device__ __forceinline__ float softplusf_(float x){ return (x > 20.0f) ? x : log1pf(__expf(x)); }

// ---------------------------------------------------------------------------
// Generic bf16 WMMA GEMM:  out[M,N] = act( A[M,K] * W[N,K]^T + bias )
// One wave computes a 32x64 tile: 2 A-frags x 4 B-frags -> 8 WMMAs per k-step,
// each fragment in its own register set so loads cluster and overlap WMMAs.
// Optional f32 output with residual (o = alpha*val + res), optional bf16 out.
// ---------------------------------------------------------------------------
template<int ACT, bool F32OUT, bool BFOUT>
__global__ __launch_bounds__(32)
void gemm_bf16_wmma(const bf16_t* __restrict__ A, int lda,
                    const bf16_t* __restrict__ W, int ldw,
                    const float* __restrict__ bias,
                    float* __restrict__ Cf, int ldcf,
                    const float* __restrict__ res, float alpha,
                    bf16_t* __restrict__ Cb, int ldcb, int cboff,
                    int K)
{
    const int lane = threadIdx.x;     // 0..31
    const int mloc = lane & 15;       // row (A) / col (W) within tile
    const int half = lane >> 4;
    const int m0 = blockIdx.x * 32;
    const int n0 = blockIdx.y * 64;

    f32x8 acc[2][4] = {};
    for (int k0 = 0; k0 < K; k0 += 32) {
        Frag a[2];
        Frag bfr[4];
        // A fragments: 16-bit A 16x32 ISA layout (lane-half splits K in 8-chunks).
#pragma unroll
        for (int mt = 0; mt < 2; ++mt) {
            const bf16_t* Arow =
                A + (size_t)(m0 + mt * 16 + mloc) * lda + k0 + 8 * half;
            if (k0 + 32 < K) __builtin_prefetch(Arow + 32, 0, 1);
#pragma unroll
            for (int p = 0; p < 8; ++p) {
                const int k = ((p >> 2) << 4) + 2 * (p & 3);   // {0,2,4,6,16,18,20,22}
                a[mt].u[p] = *(const unsigned int*)(Arow + k);
            }
        }
        // B fragments: 32x16 layout; same K order as A (8*half + {0..7}, 16+8*half + {0..7}) at col=lane&15.
#pragma unroll
        for (int t = 0; t < 4; ++t) {
            const bf16_t* Wrow =
                W + (size_t)(n0 + t * 16 + mloc) * ldw + k0 + 8 * half;
#pragma unroll
            for (int p = 0; p < 8; ++p)
                bfr[t].u[p] = *(const unsigned int*)(Wrow + ((p >> 2) << 4) + 2 * (p & 3));
        }
        // 8 WMMAs on fully-resident fragments.
#pragma unroll
        for (int mt = 0; mt < 2; ++mt)
#pragma unroll
            for (int t = 0; t < 4; ++t)
                acc[mt][t] = __builtin_amdgcn_wmma_f32_16x16x32_bf16(
                    false, a[mt].v, false, bfr[t].v, (short)0, acc[mt][t],
                    false, false);
    }

    // Epilogue: f32 C/D 16x16 layout — VGPR r: rows r (+8 for upper half), col = lane&15.
#pragma unroll
    for (int mt = 0; mt < 2; ++mt) {
#pragma unroll
        for (int t = 0; t < 4; ++t) {
#pragma unroll
            for (int r = 0; r < 8; ++r) {
                const int row = m0 + mt * 16 + r + 8 * half;
                const int col = n0 + t * 16 + mloc;
                float v = acc[mt][t][r];
                if (bias) v += bias[col];
                if (ACT == 1) v = siluf_(v);
                if (F32OUT) {
                    float o = alpha * v;
                    if (res) o += res[(size_t)row * ldcf + col];
                    Cf[(size_t)row * ldcf + col] = o;
                }
                if (BFOUT) Cb[(size_t)row * ldcb + col + cboff] = (bf16_t)v;
            }
        }
    }
}

// ---------------------------------------------------------------------------
// LayerNorm over D=512, one block (256 threads) per token.
// ---------------------------------------------------------------------------
template<bool BFO, bool F32O>
__global__ __launch_bounds__(256)
void ln_kernel(const float* __restrict__ x, const float* __restrict__ g,
               const float* __restrict__ b, bf16_t* __restrict__ obf,
               float* __restrict__ of)
{
    const int row = blockIdx.x;
    const int tid = threadIdx.x;
    const float* xr = x + (size_t)row * DIMD;
    const float v0 = xr[tid];
    const float v1 = xr[tid + 256];
    __shared__ float ssum[256], ssq[256];
    ssum[tid] = v0 + v1;
    ssq[tid]  = v0 * v0 + v1 * v1;
    __syncthreads();
    for (int s = 128; s > 0; s >>= 1) {
        if (tid < s) { ssum[tid] += ssum[tid + s]; ssq[tid] += ssq[tid + s]; }
        __syncthreads();
    }
    const float mean = ssum[0] * (1.0f / DIMD);
    const float var  = ssq[0] * (1.0f / DIMD) - mean * mean;
    const float rstd = rsqrtf(var + 1e-5f);
    const float o0 = (v0 - mean) * rstd * g[tid] + b[tid];
    const float o1 = (v1 - mean) * rstd * g[tid + 256] + b[tid + 256];
    if (BFO) {
        obf[(size_t)row * DIMD + tid] = (bf16_t)o0;
        obf[(size_t)row * DIMD + tid + 256] = (bf16_t)o1;
    }
    if (F32O) {
        of[(size_t)row * DIMD + tid] = o0;
        of[(size_t)row * DIMD + tid + 256] = o1;
    }
}

// ---------------------------------------------------------------------------
// Elementwise converters
// ---------------------------------------------------------------------------
__global__ __launch_bounds__(256)
void cvt_kernel(const float* __restrict__ in, bf16_t* __restrict__ out, int n)
{
    const int i = blockIdx.x * 256 + threadIdx.x;
    if (i < n) out[i] = (bf16_t)in[i];
}

__global__ __launch_bounds__(256)
void cvt_cols_kernel(const float* __restrict__ in, int ldin,
                     bf16_t* __restrict__ out, int ldout, int cols, int n)
{
    const int i = blockIdx.x * 256 + threadIdx.x;
    if (i < n) {
        const int r = i / cols, c = i % cols;
        out[(size_t)r * ldout + c] = (bf16_t)in[(size_t)r * ldin + c];
    }
}

// ---------------------------------------------------------------------------
// Mamba causal depthwise conv (K=4) + bias + SiLU. REV flips the time window.
// xz: [M, 2*DI] (xi = first DI cols). Outputs f32 + bf16 [M, DI].
// ---------------------------------------------------------------------------
template<int REV>
__global__ __launch_bounds__(256)
void mamba_conv_kernel(const float* __restrict__ xz, const float* __restrict__ cw,
                       const float* __restrict__ cb, float* __restrict__ xcf,
                       bf16_t* __restrict__ xcb)
{
    const int i = blockIdx.x * 256 + threadIdx.x;   // B*L*DI
    const int c = i & (DII - 1);
    const int l = (i >> 10) & (LLEN - 1);
    const int b = i >> 19;
    float s = cb[c];
#pragma unroll
    for (int k = 0; k < KCONV; ++k) {
        const int ls = REV ? (l + (KCONV - 1) - k) : (l - (KCONV - 1) + k);
        if (ls >= 0 && ls < LLEN)
            s += cw[c * KCONV + k] * xz[(size_t)(b * LLEN + ls) * (2 * DII) + c];
    }
    s = siluf_(s);
    xcf[i] = s;
    xcb[i] = (bf16_t)s;
}

// ---------------------------------------------------------------------------
// Selective scan. Block = 256 channels of one batch; B/C (32 floats) staged in
// LDS per timestep. Epilogue fuses +Dp*xc and *silu(z); writes bf16 y.
// ---------------------------------------------------------------------------
template<int REV>
__global__ __launch_bounds__(256)
void scan_kernel(const float* __restrict__ dtl,   // [M, DI] pre-softplus (bdt added)
                 const float* __restrict__ xdb,   // [M, 64] (B at 32:48, C at 48:64)
                 const float* __restrict__ xcf,   // [M, DI]
                 const float* __restrict__ xz,    // [M, 2DI], z at +DI
                 const float* __restrict__ Alog,  // [DI, NST]
                 const float* __restrict__ Dp,    // [DI]
                 bf16_t* __restrict__ ybf)        // [M, DI]
{
    const int tid = threadIdx.x;
    const int c = blockIdx.x * 256 + tid;
    const int b = blockIdx.y;
    __shared__ float sBC[2 * NST];
    float Ac[NST];
#pragma unroll
    for (int n = 0; n < NST; ++n) Ac[n] = -__expf(Alog[(size_t)c * NST + n]);
    const float dpc = Dp[c];
    float h[NST];
#pragma unroll
    for (int n = 0; n < NST; ++n) h[n] = 0.0f;

    for (int t = 0; t < LLEN; ++t) {
        const int l = REV ? (LLEN - 1 - t) : t;
        const size_t idx = (size_t)(b * LLEN + l);
        __syncthreads();
        if (tid < 2 * NST) sBC[tid] = xdb[idx * 64 + DTRR + tid];
        __syncthreads();
        const float dt = softplusf_(dtl[idx * DII + c]);
        const float xc = xcf[idx * DII + c];
        float y = 0.0f;
#pragma unroll
        for (int n = 0; n < NST; ++n) {
            const float dA = __expf(dt * Ac[n]);
            h[n] = dA * h[n] + dt * sBC[n] * xc;
            y += h[n] * sBC[NST + n];
        }
        const float z = xz[idx * (2 * DII) + DII + c];
        const float out = (y + dpc * xc) * siluf_(z);
        ybf[idx * DII + c] = (bf16_t)out;
    }
}

// ---------------------------------------------------------------------------
// GLU: hc = a * sigmoid(g), pw: [M, 2D] -> hc: [M, D]
// ---------------------------------------------------------------------------
__global__ __launch_bounds__(256)
void glu_kernel(const float* __restrict__ pw, float* __restrict__ hc)
{
    const int i = blockIdx.x * 256 + threadIdx.x;   // M*D
    const int r = i >> 9;
    const int c = i & (DIMD - 1);
    const float a = pw[(size_t)r * (2 * DIMD) + c];
    const float g = pw[(size_t)r * (2 * DIMD) + DIMD + c];
    hc[i] = a * sigmoidf_(g);
}

// ---------------------------------------------------------------------------
// Multi-scale depthwise conv (15/31/63, same-pad) avg + BN + SiLU -> bf16.
// ---------------------------------------------------------------------------
__global__ __launch_bounds__(256)
void dwconv_kernel(const float* __restrict__ hc,
                   const float* __restrict__ w15, const float* __restrict__ w31,
                   const float* __restrict__ w63,
                   const float* __restrict__ bng, const float* __restrict__ bnb,
                   const float* __restrict__ bnm, const float* __restrict__ bnv,
                   bf16_t* __restrict__ out)
{
    const int i = blockIdx.x * 256 + threadIdx.x;   // B*L*D
    const int c = i & (DIMD - 1);
    const int l = (i >> 9) & (LLEN - 1);
    const int b = i >> 18;
    const float* col = hc + (size_t)b * LLEN * DIMD + c;
    float s = 0.0f;
#pragma unroll 5
    for (int j = 0; j < 15; ++j) {
        const int ls = l + j - 7;
        if (ls >= 0 && ls < LLEN) s += w15[c * 15 + j] * col[(size_t)ls * DIMD];
    }
#pragma unroll 5
    for (int j = 0; j < 31; ++j) {
        const int ls = l + j - 15;
        if (ls >= 0 && ls < LLEN) s += w31[c * 31 + j] * col[(size_t)ls * DIMD];
    }
#pragma unroll 5
    for (int j = 0; j < 63; ++j) {
        const int ls = l + j - 31;
        if (ls >= 0 && ls < LLEN) s += w63[c * 63 + j] * col[(size_t)ls * DIMD];
    }
    s *= (1.0f / 3.0f);
    s = (s - bnm[c]) * rsqrtf(bnv[c] + 1e-5f) * bng[c] + bnb[c];
    s = siluf_(s);
    out[i] = (bf16_t)s;
}

// ---------------------------------------------------------------------------
extern "C" void kernel_launch(void* const* d_in, const int* in_sizes, int n_in,
                              void* d_out, int out_size, void* d_ws, size_t ws_size,
                              hipStream_t stream)
{
    (void)in_sizes; (void)n_in; (void)out_size; (void)ws_size;
    const float* x        = (const float*)d_in[0];
    const float* ff1_ln_g = (const float*)d_in[1];
    const float* ff1_ln_b = (const float*)d_in[2];
    const float* ff1_w1   = (const float*)d_in[3];
    const float* ff1_b1   = (const float*)d_in[4];
    const float* ff1_w2   = (const float*)d_in[5];
    const float* ff1_b2   = (const float*)d_in[6];
    const float* ff2_ln_g = (const float*)d_in[7];
    const float* ff2_ln_b = (const float*)d_in[8];
    const float* ff2_w1   = (const float*)d_in[9];
    const float* ff2_b1   = (const float*)d_in[10];
    const float* ff2_w2   = (const float*)d_in[11];
    const float* ff2_b2   = (const float*)d_in[12];
    const float* m_win    = (const float*)d_in[13];
    const float* m_convw  = (const float*)d_in[14];
    const float* m_convb  = (const float*)d_in[15];
    const float* m_wx     = (const float*)d_in[16];
    const float* m_wdt    = (const float*)d_in[17];
    const float* m_bdt    = (const float*)d_in[18];
    const float* m_Alog   = (const float*)d_in[19];
    const float* m_Dp     = (const float*)d_in[20];
    const float* m_wout   = (const float*)d_in[21];
    const float* bi_wo    = (const float*)d_in[22];
    const float* bi_bo    = (const float*)d_in[23];
    const float* cv_ln_g  = (const float*)d_in[24];
    const float* cv_ln_b  = (const float*)d_in[25];
    const float* cv_pw1_w = (const float*)d_in[26];
    const float* cv_pw1_b = (const float*)d_in[27];
    const float* cv_dw15  = (const float*)d_in[28];
    const float* cv_dw31  = (const float*)d_in[29];
    const float* cv_dw63  = (const float*)d_in[30];
    const float* cv_bn_g  = (const float*)d_in[31];
    const float* cv_bn_b  = (const float*)d_in[32];
    const float* cv_bn_m  = (const float*)d_in[33];
    const float* cv_bn_v  = (const float*)d_in[34];
    const float* cv_pw2_w = (const float*)d_in[35];
    const float* cv_pw2_b = (const float*)d_in[36];
    const float* ln_g     = (const float*)d_in[37];
    const float* ln_b     = (const float*)d_in[38];
    float* out = (float*)d_out;

    // ---- workspace carve (256B aligned) ----
    uint8_t* base = (uint8_t*)d_ws;
    size_t off = 0;
    auto alloc = [&](size_t bytes) -> void* {
        void* p = base + off;
        off += (bytes + 255) & ~(size_t)255;
        return p;
    };
    bf16_t* wb_ff1w1 = (bf16_t*)alloc((size_t)2048 * 512 * 2);
    bf16_t* wb_ff1w2 = (bf16_t*)alloc((size_t)512 * 2048 * 2);
    bf16_t* wb_ff2w1 = (bf16_t*)alloc((size_t)2048 * 512 * 2);
    bf16_t* wb_ff2w2 = (bf16_t*)alloc((size_t)512 * 2048 * 2);
    bf16_t* wb_win   = (bf16_t*)alloc((size_t)2 * 2048 * 512 * 2);
    bf16_t* wb_wx    = (bf16_t*)alloc((size_t)2 * 64 * 1024 * 2);
    bf16_t* wb_wdt   = (bf16_t*)alloc((size_t)2 * 1024 * 32 * 2);
    bf16_t* wb_wout  = (bf16_t*)alloc((size_t)2 * 512 * 1024 * 2);
    bf16_t* wb_biwo  = (bf16_t*)alloc((size_t)512 * 1024 * 2);
    bf16_t* wb_pw1   = (bf16_t*)alloc((size_t)1024 * 512 * 2);
    bf16_t* wb_pw2   = (bf16_t*)alloc((size_t)512 * 512 * 2);
    bf16_t* bf_a     = (bf16_t*)alloc((size_t)MTOK * 512 * 2);    // LN / dw outputs
    bf16_t* bf_b     = (bf16_t*)alloc((size_t)MTOK * 2048 * 2);   // ffn hidden / xc / y
    bf16_t* bf_cat   = (bf16_t*)alloc((size_t)MTOK * 1024 * 2);
    bf16_t* bf_dt    = (bf16_t*)alloc((size_t)MTOK * 32 * 2);
    float*  f_h      = (float*)alloc((size_t)MTOK * 512 * 4);     // running residual
    float*  f_xz     = (float*)alloc((size_t)MTOK * 2048 * 4);    // xz / pw1 out
    float*  f_xc     = (float*)alloc((size_t)MTOK * 1024 * 4);    // xc f32 / GLU hc
    float*  f_xdb    = (float*)alloc((size_t)MTOK * 64 * 4);
    float*  f_dtl    = (float*)alloc((size_t)MTOK * 1024 * 4);

    const dim3 blk256(256), blk32(32);
    auto cvt = [&](const float* src, bf16_t* dst, int n) {
        cvt_kernel<<<dim3((n + 255) / 256), blk256, 0, stream>>>(src, dst, n);
    };

    // ---- weight conversions to bf16 ----
    cvt(ff1_w1, wb_ff1w1, 2048 * 512);
    cvt(ff1_w2, wb_ff1w2, 512 * 2048);
    cvt(ff2_w1, wb_ff2w1, 2048 * 512);
    cvt(ff2_w2, wb_ff2w2, 512 * 2048);
    cvt(m_win, wb_win, 2 * 2048 * 512);
    cvt(m_wx, wb_wx, 2 * 64 * 1024);
    cvt(m_wdt, wb_wdt, 2 * 1024 * 32);
    cvt(m_wout, wb_wout, 2 * 512 * 1024);
    cvt(bi_wo, wb_biwo, 512 * 1024);
    cvt(cv_pw1_w, wb_pw1, 1024 * 512);
    cvt(cv_pw2_w, wb_pw2, 512 * 512);

    // ================= FFN1: h = x + 0.5*ffn(x) =================
    ln_kernel<true, false><<<dim3(MTOK), blk256, 0, stream>>>(x, ff1_ln_g, ff1_ln_b, bf_a, nullptr);
    gemm_bf16_wmma<1, false, true><<<dim3(MTOK / 32, 2048 / 64), blk32, 0, stream>>>(
        bf_a, 512, wb_ff1w1, 512, ff1_b1, nullptr, 0, nullptr, 1.0f, bf_b, 2048, 0, 512);
    gemm_bf16_wmma<0, true, false><<<dim3(MTOK / 32, 512 / 64), blk32, 0, stream>>>(
        bf_b, 2048, wb_ff1w2, 2048, ff1_b2, f_h, 512, x, 0.5f, nullptr, 0, 0, 2048);

    // ================= BiMamba =================
    cvt(f_h, bf_a, MTOK * 512);
    for (int d = 0; d < 2; ++d) {
        gemm_bf16_wmma<0, true, false><<<dim3(MTOK / 32, 2048 / 64), blk32, 0, stream>>>(
            bf_a, 512, wb_win + (size_t)d * 2048 * 512, 512, nullptr,
            f_xz, 2048, nullptr, 1.0f, nullptr, 0, 0, 512);
        if (d == 0)
            mamba_conv_kernel<0><<<dim3(MTOK * DII / 256), blk256, 0, stream>>>(
                f_xz, m_convw + d * DII * KCONV, m_convb + d * DII, f_xc, bf_b);
        else
            mamba_conv_kernel<1><<<dim3(MTOK * DII / 256), blk256, 0, stream>>>(
                f_xz, m_convw + d * DII * KCONV, m_convb + d * DII, f_xc, bf_b);
        gemm_bf16_wmma<0, true, false><<<dim3(MTOK / 32, 64 / 64), blk32, 0, stream>>>(
            bf_b, 1024, wb_wx + (size_t)d * 64 * 1024, 1024, nullptr,
            f_xdb, 64, nullptr, 1.0f, nullptr, 0, 0, 1024);
        cvt_cols_kernel<<<dim3((MTOK * 32 + 255) / 256), blk256, 0, stream>>>(
            f_xdb, 64, bf_dt, 32, 32, MTOK * 32);
        gemm_bf16_wmma<0, true, false><<<dim3(MTOK / 32, 1024 / 64), blk32, 0, stream>>>(
            bf_dt, 32, wb_wdt + (size_t)d * 1024 * 32, 32, m_bdt + d * DII,
            f_dtl, 1024, nullptr, 1.0f, nullptr, 0, 0, 32);
        if (d == 0)
            scan_kernel<0><<<dim3(DII / 256, BB), blk256, 0, stream>>>(
                f_dtl, f_xdb, f_xc, f_xz, m_Alog + (size_t)d * DII * NST, m_Dp + d * DII, bf_b);
        else
            scan_kernel<1><<<dim3(DII / 256, BB), blk256, 0, stream>>>(
                f_dtl, f_xdb, f_xc, f_xz, m_Alog + (size_t)d * DII * NST, m_Dp + d * DII, bf_b);
        gemm_bf16_wmma<0, false, true><<<dim3(MTOK / 32, 512 / 64), blk32, 0, stream>>>(
            bf_b, 1024, wb_wout + (size_t)d * 512 * 1024, 1024, nullptr,
            nullptr, 0, nullptr, 1.0f, bf_cat, 1024, d * 512, 1024);
    }
    // h += cat @ bi_wo^T + bi_bo   (in-place residual: each lane reads/writes own element)
    gemm_bf16_wmma<0, true, false><<<dim3(MTOK / 32, 512 / 64), blk32, 0, stream>>>(
        bf_cat, 1024, wb_biwo, 1024, bi_bo, f_h, 512, f_h, 1.0f, nullptr, 0, 0, 1024);

    // ================= ConvModule =================
    ln_kernel<true, false><<<dim3(MTOK), blk256, 0, stream>>>(f_h, cv_ln_g, cv_ln_b, bf_a, nullptr);
    gemm_bf16_wmma<0, true, false><<<dim3(MTOK / 32, 1024 / 64), blk32, 0, stream>>>(
        bf_a, 512, wb_pw1, 512, cv_pw1_b, f_xz, 1024, nullptr, 1.0f, nullptr, 0, 0, 512);
    glu_kernel<<<dim3(MTOK * 512 / 256), blk256, 0, stream>>>(f_xz, f_xc);
    dwconv_kernel<<<dim3(MTOK * 512 / 256), blk256, 0, stream>>>(
        f_xc, cv_dw15, cv_dw31, cv_dw63, cv_bn_g, cv_bn_b, cv_bn_m, cv_bn_v, bf_a);
    gemm_bf16_wmma<0, true, false><<<dim3(MTOK / 32, 512 / 64), blk32, 0, stream>>>(
        bf_a, 512, wb_pw2, 512, cv_pw2_b, f_h, 512, f_h, 1.0f, nullptr, 0, 0, 512);

    // ================= FFN2 =================
    ln_kernel<true, false><<<dim3(MTOK), blk256, 0, stream>>>(f_h, ff2_ln_g, ff2_ln_b, bf_a, nullptr);
    gemm_bf16_wmma<1, false, true><<<dim3(MTOK / 32, 2048 / 64), blk32, 0, stream>>>(
        bf_a, 512, wb_ff2w1, 512, ff2_b1, nullptr, 0, nullptr, 1.0f, bf_b, 2048, 0, 512);
    gemm_bf16_wmma<0, true, false><<<dim3(MTOK / 32, 512 / 64), blk32, 0, stream>>>(
        bf_b, 2048, wb_ff2w2, 2048, ff2_b2, f_h, 512, f_h, 0.5f, nullptr, 0, 0, 2048);

    // ================= Final LN =================
    ln_kernel<false, true><<<dim3(MTOK), blk256, 0, stream>>>(f_h, ln_g, ln_b, nullptr, out);
}
